// GCN_Only_Model_26001732010443
// MI455X (gfx1250) — matrix operand order repaired and checked by reading, not compiled, not run on hardware
//
#include <hip/hip_runtime.h>
#include <hip/hip_bf16.h>

#define B_   32
#define WIN_ 100
#define N_   2048
#define K_   20
#define E_   64

typedef __attribute__((ext_vector_type(16))) _Float16 v16h;
typedef __attribute__((ext_vector_type(8)))  _Float16 v8h;
typedef __attribute__((ext_vector_type(8)))  float    v8f;

#define NEG_INF (-__builtin_inff())

// ---------------------------------------------------------------------------
// Kernel 1: q/k projections  q[b,n,e] = sum_w x[b,w,n]*Wq[w,e] + bq[e]  -> f16
// ---------------------------------------------------------------------------
__global__ void qk_kernel(const float* __restrict__ x,
                          const float* __restrict__ Wq, const float* __restrict__ bq,
                          const float* __restrict__ Wk, const float* __restrict__ bk,
                          _Float16* __restrict__ qh, _Float16* __restrict__ kh) {
  int t = blockIdx.x * blockDim.x + threadIdx.x;   // over B*N*E
  int e = t & (E_ - 1);
  int n = (t >> 6) & (N_ - 1);
  int b = t >> 17;
  const float* xp = x + (size_t)b * WIN_ * N_ + n;
  float q = bq[e], k = bk[e];
#pragma unroll 4
  for (int w = 0; w < WIN_; ++w) {
    float xv = xp[(size_t)w * N_];
    q = fmaf(xv, Wq[w * E_ + e], q);
    k = fmaf(xv, Wk[w * E_ + e], k);
  }
  qh[t] = (_Float16)q;
  kh[t] = (_Float16)k;
}

// ---------------------------------------------------------------------------
// Kernel 2: xw[b,n,w] = sum_w' x[b,w',n] * Wg[w',w]
// ---------------------------------------------------------------------------
__global__ void xw_kernel(const float* __restrict__ x, const float* __restrict__ Wg,
                          float* __restrict__ xw) {
  int t = blockIdx.x * blockDim.x + threadIdx.x;   // over B*N*WIN
  int w = t % WIN_;
  int n = (t / WIN_) % N_;
  int b = t / (WIN_ * N_);
  const float* xp = x + (size_t)b * WIN_ * N_ + n;
  float acc = 0.f;
#pragma unroll 4
  for (int w2 = 0; w2 < WIN_; ++w2)
    acc = fmaf(xp[(size_t)w2 * N_], Wg[w2 * WIN_ + w], acc);
  xw[t] = acc;
}

// ---------------------------------------------------------------------------
// WMMA fragment loaders (wave32), contiguous k = 16h + s order used for both A and B
// A 16x32 f16: lane<16 holds K k0..k0+15 with k0=kbase;
//              lane>=16 same with k0=kbase+16.
// B 32x16 f16: lanes 0-15 hold K kbase..kbase+15 of col=lane;
//              lanes 16-31 hold K kbase+16..kbase+31 of col=lane-16.
// ---------------------------------------------------------------------------
__device__ __forceinline__ v16h load_frag_a(const _Float16* __restrict__ feat,
                                            int kbase, int lane) {
  int m  = lane & 15;
  int k0 = kbase + ((lane & 16) ? 16 : 0);
  const _Float16* p = feat + m * E_ + k0;
  v8h lo = *(const v8h*)(p);        // K k0 .. k0+7
  v8h hi = *(const v8h*)(p + 8);    // K k0+8 .. k0+15
  v16h r;
#pragma unroll
  for (int i = 0; i < 8; ++i) { r[i] = lo[i]; r[i + 8] = hi[i]; }
  return r;
}

__device__ __forceinline__ v16h load_frag_b(const _Float16* __restrict__ feat,
                                            int kbase, int lane) {
  int ncol = lane & 15;
  int k0   = kbase + ((lane & 16) ? 16 : 0);
  return *(const v16h*)(feat + ncol * E_ + k0);   // 16 contiguous halves
}

// Scale 1/sqrt(E), mask the diagonal, spill C tile to LDS row-major.
__device__ __forceinline__ void store_tile(float* __restrict__ sc, int STRIDE,
                                           int row_base, int ct, v8f c,
                                           int ncol, int m_off) {
  int col = ct * 16 + ncol;
#pragma unroll
  for (int r = 0; r < 8; ++r) {
    int m = m_off + r;                             // C layout: M = r + 8*(lane>=16)
    float v = c[r] * 0.125f;                       // 1/sqrt(E)
    if (row_base + m == col) v = NEG_INF;          // mask diagonal
    sc[m * STRIDE + col] = v;
  }
}

// ---------------------------------------------------------------------------
// Kernel 3 (fused): adj tile [16 x 2048] via v_wmma_f32_16x16x32_f16 into LDS,
// then iterative top-20 per row.  Block = 256 threads (8 waves); block owns
// (batch b, 16-row tile).  B fragments use a 2-deep register pipeline with
// two independent buffer sets (no rotation movs; wmma waits loadcnt<=4).
// ---------------------------------------------------------------------------
__global__ void adj_topk_kernel(const _Float16* __restrict__ qh,
                                const _Float16* __restrict__ kh,
                                int* __restrict__ idx_out) {
  extern __shared__ float sc[];                    // [16][N_+4]
  const int STRIDE = N_ + 4;
  const int NT     = N_ / 16;                      // 128 column tiles
  int b        = blockIdx.x >> 7;                  // / (N_/16 == 128)
  int rt       = blockIdx.x & 127;
  int row_base = rt * 16;
  int lane = threadIdx.x & 31;
  int wid  = threadIdx.x >> 5;

  const _Float16* qrow = qh + ((size_t)b * N_ + row_base) * E_;
  v16h a0 = load_frag_a(qrow, 0,  lane);           // K 0..31
  v16h a1 = load_frag_a(qrow, 32, lane);           // K 32..63

  int ncol  = lane & 15;
  int m_off = (lane & 16) ? 8 : 0;

  const _Float16* kbp = kh + (size_t)b * N_ * E_;

  // Prologue: wave owns tiles wid, wid+8, ..., wid+120 (16 tiles).
  // Two buffer sets cover tiles ct (A-set) and ct+8 (B-set).
  v16h A0 = load_frag_b(kbp + (size_t)wid * 16 * E_, 0,  lane);
  v16h A1 = load_frag_b(kbp + (size_t)wid * 16 * E_, 32, lane);
  v16h B0 = load_frag_b(kbp + (size_t)(wid + 8) * 16 * E_, 0,  lane);
  v16h B1 = load_frag_b(kbp + (size_t)(wid + 8) * 16 * E_, 32, lane);

  for (int ct = wid; ct < NT; ct += 16) {          // 8 iterations, 2 tiles each
    // ---- tile ct: consume A-set, refill A-set from ct+16 -------------------
    {
      int ctn = (ct + 16 < NT) ? ct + 16 : ct;     // clamp on last iteration
      const _Float16* kn = kbp + (size_t)ctn * 16 * E_;
      v8f c = {};
      c = __builtin_amdgcn_wmma_f32_16x16x32_f16(false, a0, false, A0,
                                                 (short)0, c, false, false);
      c = __builtin_amdgcn_wmma_f32_16x16x32_f16(false, a1, false, A1,
                                                 (short)0, c, false, false);
      A0 = load_frag_b(kn, 0,  lane);              // issue behind the wmma
      A1 = load_frag_b(kn, 32, lane);
      store_tile(sc, STRIDE, row_base, ct, c, ncol, m_off);
    }
    // ---- tile ct+8: consume B-set, refill B-set from ct+24 -----------------
    {
      int ct2 = ct + 8;
      int ctn = (ct2 + 16 < NT) ? ct2 + 16 : ct2;
      const _Float16* kn = kbp + (size_t)ctn * 16 * E_;
      v8f c = {};
      c = __builtin_amdgcn_wmma_f32_16x16x32_f16(false, a0, false, B0,
                                                 (short)0, c, false, false);
      c = __builtin_amdgcn_wmma_f32_16x16x32_f16(false, a1, false, B1,
                                                 (short)0, c, false, false);
      B0 = load_frag_b(kn, 0,  lane);
      B1 = load_frag_b(kn, 32, lane);
      store_tile(sc, STRIDE, row_base, ct2, c, ncol, m_off);
    }
  }
  __syncthreads();

  // Phase 2: wave `wid` extracts top-K of rows 2*wid and 2*wid+1.
  for (int rr = 0; rr < 2; ++rr) {
    int row   = wid * 2 + rr;
    int row_g = row_base + row;
    float* rp = sc + row * STRIDE;
    for (int sel = 0; sel < K_; ++sel) {
      float best = NEG_INF;
      int   bi   = lane;
#pragma unroll 8
      for (int cidx = lane; cidx < N_; cidx += 32) {
        float v = rp[cidx];
        if (v > best) { best = v; bi = cidx; }
      }
#pragma unroll
      for (int off = 16; off >= 1; off >>= 1) {    // wave32 reduction
        float ov = __shfl_xor(best, off, 32);
        int   oi = __shfl_xor(bi, off, 32);
        if (ov > best || (ov == best && oi < bi)) { best = ov; bi = oi; }
      }
      if (lane == 0) {
        idx_out[((size_t)b * N_ + row_g) * K_ + sel] = bi;
        rp[bi] = NEG_INF;                          // extract
      }
      __builtin_amdgcn_wave_barrier();             // DS ops in-order within wave
    }
  }
}

// ---------------------------------------------------------------------------
// Kernels 4-9: GCN normalization + scatter aggregation
// ---------------------------------------------------------------------------
__global__ void deg_init_kernel(float* __restrict__ deg) {
  int t = blockIdx.x * blockDim.x + threadIdx.x;   // B*N
  deg[t] = 1.0f;                                   // self loop
}

__global__ void deg_count_kernel(const int* __restrict__ idx, float* __restrict__ deg) {
  int t = blockIdx.x * blockDim.x + threadIdx.x;   // B*N*K
  int b = t / (N_ * K_);
  atomicAdd(&deg[b * N_ + idx[t]], 1.0f);
}

__global__ void dinv_kernel(float* __restrict__ deg) {
  int t = blockIdx.x * blockDim.x + threadIdx.x;   // B*N  (deg >= 1 always)
  deg[t] = rsqrtf(deg[t]);
}

__global__ void h_init_kernel(const float* __restrict__ xw, const float* __restrict__ dinv,
                              float* __restrict__ h) {
  int t = blockIdx.x * blockDim.x + threadIdx.x;   // B*N*WIN (self-loop term)
  float d = dinv[t / WIN_];
  h[t] = d * d * xw[t];
}

__global__ void scatter_kernel(const float* __restrict__ xw, const float* __restrict__ dinv,
                               const int* __restrict__ idx, float* __restrict__ h) {
  int t = blockIdx.x * blockDim.x + threadIdx.x;   // B*N*K*WIN
  int e   = t / WIN_;
  int w   = t - e * WIN_;
  int src = (e / K_) & (N_ - 1);
  int b   = e / (N_ * K_);
  int tgt = idx[e];
  float s = dinv[b * N_ + src] * dinv[b * N_ + tgt];
  atomicAdd(&h[((size_t)b * N_ + tgt) * WIN_ + w],
            s * xw[((size_t)b * N_ + src) * WIN_ + w]);
}

__global__ void out_kernel(const float* __restrict__ h, const float* __restrict__ bg,
                           float* __restrict__ out) {
  int t = blockIdx.x * blockDim.x + threadIdx.x;   // B*WIN*N, n fastest
  int n = t & (N_ - 1);
  int w = (t >> 11) % WIN_;
  int b = t / (WIN_ * N_);
  out[t] = h[((size_t)b * N_ + n) * WIN_ + w] + bg[w];
}

// ---------------------------------------------------------------------------
extern "C" void kernel_launch(void* const* d_in, const int* in_sizes, int n_in,
                              void* d_out, int out_size, void* d_ws, size_t ws_size,
                              hipStream_t stream) {
  const float* x  = (const float*)d_in[0];
  const float* Wq = (const float*)d_in[1];
  const float* bq = (const float*)d_in[2];
  const float* Wk = (const float*)d_in[3];
  const float* bk = (const float*)d_in[4];
  const float* Wg = (const float*)d_in[5];
  const float* bg = (const float*)d_in[6];
  float* out = (float*)d_out;
  (void)in_sizes; (void)n_in; (void)out_size; (void)ws_size;

  char* p = (char*)d_ws;
  auto take = [&](size_t bytes) { char* r = p; p += (bytes + 255) & ~(size_t)255; return r; };
  _Float16* qh  = (_Float16*)take((size_t)B_ * N_ * E_ * 2);
  _Float16* kh  = (_Float16*)take((size_t)B_ * N_ * E_ * 2);
  int*      idx = (int*)     take((size_t)B_ * N_ * K_ * 4);
  float*    xw  = (float*)   take((size_t)B_ * N_ * WIN_ * 4);
  float*    deg = (float*)   take((size_t)B_ * N_ * 4);
  float*    h   = (float*)   take((size_t)B_ * N_ * WIN_ * 4);

  qk_kernel     <<<(B_ * N_ * E_) / 256,        256, 0, stream>>>(x, Wq, bq, Wk, bk, qh, kh);
  xw_kernel     <<<(B_ * N_ * WIN_) / 256,      256, 0, stream>>>(x, Wg, xw);

  size_t lds_bytes = (size_t)16 * (N_ + 4) * sizeof(float);   // ~128 KB of 320 KB/WGP
  adj_topk_kernel<<<B_ * (N_ / 16), 256, lds_bytes, stream>>>(qh, kh, idx);

  deg_init_kernel <<<(B_ * N_) / 256,           256, 0, stream>>>(deg);
  deg_count_kernel<<<(B_ * N_ * K_) / 256,      256, 0, stream>>>(idx, deg);
  dinv_kernel     <<<(B_ * N_) / 256,           256, 0, stream>>>(deg);
  h_init_kernel   <<<(B_ * N_ * WIN_) / 256,    256, 0, stream>>>(xw, deg, h);
  scatter_kernel  <<<(B_ * N_ * K_ * WIN_) / 256, 256, 0, stream>>>(xw, deg, idx, h);
  out_kernel      <<<(B_ * WIN_ * N_) / 256,    256, 0, stream>>>(h, bg, out);
}
